// QAttention_75969381532133
// MI455X (gfx1250) — matrix operand order repaired and checked by reading, not compiled, not run on hardware
//
#include <hip/hip_runtime.h>
#include <hip/hip_bf16.h>
#include <math.h>

// ---------------- CDNA5 WMMA types ----------------
typedef __attribute__((ext_vector_type(16))) __bf16 v16bf;
typedef __attribute__((ext_vector_type(8)))  float  v8f;

#define B_    2
#define DIM   128
#define HH    128
#define WW    128
#define HW    (HH*WW)       // 16384
#define PTOT  (B_*HW)       // 32768
#define HEADS 4
#define C3    (3*DIM)       // 384
#define EPSF  1e-12f

__device__ __forceinline__ unsigned short f2bf(float f){
  unsigned int u = __float_as_uint(f);
  unsigned int r = (u + 0x7FFFu + ((u >> 16) & 1u)) >> 16;   // RNE
  return (unsigned short)r;
}

__device__ __forceinline__ v8f wmma_bf16(v16bf a, v16bf b, v8f c){
  // (neg_a, A, neg_b, B, c_mod, C, reuse_a, reuse_b) -> v_wmma_f32_16x16x32_bf16
  return __builtin_amdgcn_wmma_f32_16x16x32_bf16(false, a, false, b, (short)0, c, false, false);
}

// A-fragment contiguous K order (matches B loads): element e of a lane's 16-elem fragment -> K index
__device__ __forceinline__ int kA_map(int lane, int e){
  int k = e;
  if (lane >= 16) k += 16; // upper lane half offset
  return k;
}

// Hamilton product block matrix: row_r=[r,-i,-j,-k]; row_i=[i,r,-k,j]; row_j=[j,k,r,-i]; row_k=[k,-j,i,r]
__device__ __forceinline__ void qsel(int rb, int cb, int& src, float& sgn){
  const int   stab[16] = {0,1,2,3, 1,0,3,2, 2,3,0,1, 3,2,1,0};
  const float gtab[16] = {1.f,-1.f,-1.f,-1.f, 1.f,1.f,-1.f,1.f, 1.f,1.f,1.f,-1.f, 1.f,-1.f,1.f,1.f};
  src = stab[rb*4+cb]; sgn = gtab[rb*4+cb];
}

// ---------------- weight packing into WMMA A-fragment order ----------------
// Layout: [oct][kt][lane(32)][e(16)] bf16, oc = oct*16 + (lane&15), k = kt*32 + kA_map(lane,e)
__global__ void k_pack_1x1(const float* __restrict__ r, const float* __restrict__ i_,
                           const float* __restrict__ j_, const float* __restrict__ k_,
                           unsigned short* __restrict__ Wp,
                           int n_oct, int n_kt, int rowblk, int colblk){
  int idx = blockIdx.x * 256 + threadIdx.x;
  if (idx >= n_oct * n_kt * 512) return;
  int e    = idx & 15;
  int lane = (idx >> 4) & 31;
  int kt   = (idx >> 9) % n_kt;
  int ot   = (idx >> 9) / n_kt;
  int oc = ot * 16 + (lane & 15);
  int k  = kt * 32 + kA_map(lane, e);
  int rb = oc / rowblk, rr = oc % rowblk;
  int cb = k  / colblk, cc = k  % colblk;
  int src; float sgn; qsel(rb, cb, src, sgn);
  const float* bl = (src == 0) ? r : (src == 1) ? i_ : (src == 2) ? j_ : k_;
  Wp[idx] = f2bf(sgn * bl[rr * colblk + cc]);
}

// 3x3: layout [kk=ky*3+kx][oct(24)][ict(12)][lane][e], blocks are (96,96,3,3)
__global__ void k_pack_3x3(const float* __restrict__ r, const float* __restrict__ i_,
                           const float* __restrict__ j_, const float* __restrict__ k_,
                           unsigned short* __restrict__ Wp){
  int idx = blockIdx.x * 256 + threadIdx.x;
  if (idx >= 9 * 24 * 12 * 512) return;
  int e    = idx & 15;
  int lane = (idx >> 4) & 31;
  int t    = idx >> 9;
  int it   = t % 12; t /= 12;
  int ot   = t % 24;
  int kk   = t / 24;                 // ky*3+kx
  int oc = ot * 16 + (lane & 15);
  int ic = it * 32 + kA_map(lane, e);
  int rb = oc / 96, rr = oc % 96;
  int cb = ic / 96, cc = ic % 96;
  int src; float sgn; qsel(rb, cb, src, sgn);
  const float* bl = (src == 0) ? r : (src == 1) ? i_ : (src == 2) ? j_ : k_;
  Wp[idx] = f2bf(sgn * bl[(rr * 96 + cc) * 9 + kk]);
}

// ---------------- NCHW f32 -> channels-last bf16 ----------------
__global__ void k_x2bf(const float* __restrict__ x, unsigned short* __restrict__ Xbf){
  int idx = blockIdx.x * 256 + threadIdx.x;       // PTOT*DIM
  int c = idx & (DIM - 1);
  int P = idx >> 7;
  int n = P >> 14, p = P & (HW - 1);
  Xbf[idx] = f2bf(x[(n * DIM + c) * HW + p]);
}

// ---------------- 1x1 conv (qkv): Y[col][384] bf16 = W(384x128) @ X[col][128] + b ----------------
__global__ void k_conv_qkv(const unsigned short* __restrict__ Xbf,
                           const unsigned short* __restrict__ Wp,
                           const float* __restrict__ bias,
                           unsigned short* __restrict__ Y){
  int lane = threadIdx.x & 31;
  int tile = blockIdx.x * 8 + (threadIdx.x >> 5);   // 24*2048 tiles
  int oct = tile >> 11;
  int pt  = tile & 2047;
  int col = pt * 16 + (lane & 15);
  int khalf = (lane < 16) ? 0 : 16;                 // B-fragment lane-half K offset
  v8f c = {};
  const unsigned short* ap = Wp  + (oct * 4) * 512 + lane * 16;
  const unsigned short* bp = Xbf + col * DIM + khalf;
#pragma unroll
  for (int kt = 0; kt < 4; ++kt){
    v16bf a = *(const v16bf*)(ap + kt * 512);
    v16bf b = *(const v16bf*)(bp + kt * 32);
    c = wmma_bf16(a, b, c);
  }
  int mbase = oct * 16 + ((lane < 16) ? 0 : 8);     // C layout: lane<16 M=r, else M=r+8
  unsigned short* yp = Y + col * C3 + mbase;
#pragma unroll
  for (int r = 0; r < 8; r += 2){
    unsigned int u = (unsigned int)f2bf(c[r]   + bias[mbase + r]) |
                    ((unsigned int)f2bf(c[r+1] + bias[mbase + r + 1]) << 16);
    *(unsigned int*)(yp + r) = u;
  }
}

// ---------------- 3x3 conv (dw): implicit GEMM, K = 384*9, out f32 channels-last ----------------
__global__ void k_conv_dw(const unsigned short* __restrict__ Xbf,   // QKV1 [P][384] bf16
                          const unsigned short* __restrict__ Wp,
                          const float* __restrict__ bias,
                          float* __restrict__ Y){                    // [P][384] f32
  int lane = threadIdx.x & 31;
  int tile = blockIdx.x * 8 + (threadIdx.x >> 5);   // 24*2048 tiles
  int oct = tile >> 11;
  int pt  = tile & 2047;
  int n   = pt >> 10;
  int rem = pt & 1023;
  int y   = rem >> 3;
  int xb  = (rem & 7) << 4;
  int xx  = xb + (lane & 15);
  int khalf = (lane < 16) ? 0 : 16;
  v8f c = {};
#pragma unroll
  for (int ky = 0; ky < 3; ++ky){
    int yy = y + ky - 1;
    if (yy < 0 || yy >= HH) continue;               // uniform across the wave
#pragma unroll
    for (int kx = 0; kx < 3; ++kx){
      int xs = xx + kx - 1;
      bool ok = (xs >= 0) && (xs < WW);             // per-lane halo mask
      const unsigned short* bp = Xbf + (size_t)(n * HW + yy * WW + (ok ? xs : 0)) * C3 + khalf;
      const unsigned short* ap = Wp + (size_t)(((ky * 3 + kx) * 24 + oct) * 12) * 512 + lane * 16;
#pragma unroll
      for (int it = 0; it < 12; ++it){
        v16bf a = *(const v16bf*)(ap + it * 512);
        v16bf b = {};
        if (ok) b = *(const v16bf*)(bp + it * 32);
        c = wmma_bf16(a, b, c);
      }
    }
  }
  int mbase = oct * 16 + ((lane < 16) ? 0 : 8);
  int col = n * HW + y * WW + xx;
  float* yp = Y + (size_t)col * C3 + mbase;
  float4 o0, o1;
  o0.x = c[0] + bias[mbase+0]; o0.y = c[1] + bias[mbase+1];
  o0.z = c[2] + bias[mbase+2]; o0.w = c[3] + bias[mbase+3];
  o1.x = c[4] + bias[mbase+4]; o1.y = c[5] + bias[mbase+5];
  o1.z = c[6] + bias[mbase+6]; o1.w = c[7] + bias[mbase+7];
  *(float4*)(yp)     = o0;
  *(float4*)(yp + 4) = o1;
}

// ---------------- L2 norms of q,k channels over spatial axis ----------------
__global__ void k_norms(const float* __restrict__ D, float* __restrict__ norms){
  // blockIdx = ((b*2 + qk)*4 + h); 256 threads: c = tid&31, seg = tid>>5
  int b  = blockIdx.x >> 3;
  int qk = (blockIdx.x >> 2) & 1;
  int h  = blockIdx.x & 3;
  int choff = qk * 128 + h * 32;
  int c = threadIdx.x & 31, seg = threadIdx.x >> 5;
  float s = 0.f;
  for (int p = seg * 2048; p < (seg + 1) * 2048; ++p){
    float v = D[(size_t)((b << 14) + p) * C3 + choff + c];
    s += v * v;
  }
  __shared__ float red[256];
  red[threadIdx.x] = s;
  __syncthreads();
  if (threadIdx.x < 32){
    float t = 0.f;
#pragma unroll
    for (int g = 0; g < 8; ++g) t += red[g * 32 + threadIdx.x];
    norms[blockIdx.x * 32 + threadIdx.x] = sqrtf(t);
  }
}

// ---------------- attn = softmax(qn @ kn^T * temp) : one block per (b,h) ----------------
__global__ void k_attn(const float* __restrict__ D, const float* __restrict__ norms,
                       const float* __restrict__ temperature, float* __restrict__ AT){
  int b = blockIdx.x >> 2;
  int h = blockIdx.x & 3;
  __shared__ float qs[64 * 32];
  __shared__ float ks[64 * 32];
  __shared__ float at[1024];
  int pairbase = threadIdx.x * 4;
  int cq  = pairbase >> 5;
  int ck0 = pairbase & 31;
  float acc0 = 0.f, acc1 = 0.f, acc2 = 0.f, acc3 = 0.f;
  for (int p0 = 0; p0 < HW; p0 += 64){
#pragma unroll
    for (int i = 0; i < 8; ++i){
      int idx = threadIdx.x + i * 256;              // 2048 elems
      int row = idx >> 5, cc = idx & 31;
      size_t gp = (size_t)((b << 14) + p0 + row) * C3;
      qs[idx] = D[gp + h * 32 + cc];
      ks[idx] = D[gp + 128 + h * 32 + cc];
    }
    __syncthreads();
#pragma unroll 8
    for (int pp = 0; pp < 64; ++pp){
      float qv = qs[pp * 32 + cq];
      const float* kr = &ks[pp * 32 + ck0];
      acc0 += qv * kr[0]; acc1 += qv * kr[1];
      acc2 += qv * kr[2]; acc3 += qv * kr[3];
    }
    __syncthreads();
  }
  float tsc = temperature[h];
  float qn  = fmaxf(norms[((b * 2 + 0) * 4 + h) * 32 + cq], EPSF);
  const float* knp = norms + ((b * 2 + 1) * 4 + h) * 32;
  at[pairbase + 0] = acc0 * tsc / (qn * fmaxf(knp[ck0 + 0], EPSF));
  at[pairbase + 1] = acc1 * tsc / (qn * fmaxf(knp[ck0 + 1], EPSF));
  at[pairbase + 2] = acc2 * tsc / (qn * fmaxf(knp[ck0 + 2], EPSF));
  at[pairbase + 3] = acc3 * tsc / (qn * fmaxf(knp[ck0 + 3], EPSF));
  __syncthreads();
  if (threadIdx.x < 32){
    int row = threadIdx.x;
    float mx = -3.4e38f;
#pragma unroll
    for (int k = 0; k < 32; ++k) mx = fmaxf(mx, at[row * 32 + k]);
    float sm = 0.f;
#pragma unroll
    for (int k = 0; k < 32; ++k) sm += __expf(at[row * 32 + k] - mx);
    float inv = 1.f / sm;
    float* dst = AT + (size_t)blockIdx.x * 1024 + row * 32;
#pragma unroll
    for (int k = 0; k < 32; ++k) dst[k] = __expf(at[row * 32 + k] - mx) * inv;
  }
}

// ---------------- out = attn @ v  -> bf16 channels-last [P][128] ----------------
__global__ void k_attnout(const float* __restrict__ D, const float* __restrict__ AT,
                          unsigned short* __restrict__ AO){
  int bh    = blockIdx.x >> 11;                     // 8 (b,h) groups x 2048 chunks
  int chunk = blockIdx.x & 2047;
  int b = bh >> 2, h = bh & 3;
  __shared__ float at[1024];
#pragma unroll
  for (int i = 0; i < 4; ++i)
    at[threadIdx.x * 4 + i] = AT[(size_t)bh * 1024 + threadIdx.x * 4 + i];
  __syncthreads();
  int c  = threadIdx.x & 31;
  int pl = threadIdx.x >> 5;
  int p  = chunk * 8 + pl;
  const float* vr = D + (size_t)((b << 14) + p) * C3 + 256 + h * 32;
  float s = 0.f;
#pragma unroll
  for (int k = 0; k < 32; ++k) s += at[c * 32 + k] * vr[k];
  AO[(size_t)((b << 14) + p) * DIM + h * 32 + c] = f2bf(s);
}

// ---------------- 1x1 conv (po): d_out (NCHW f32) = W(128x128) @ AO + b ----------------
__global__ void k_conv_po(const unsigned short* __restrict__ AO,
                          const unsigned short* __restrict__ Wp,
                          const float* __restrict__ bias,
                          float* __restrict__ out){
  int lane = threadIdx.x & 31;
  int tile = blockIdx.x * 8 + (threadIdx.x >> 5);   // 8*2048 tiles
  int oct = tile >> 11;
  int pt  = tile & 2047;
  int col = pt * 16 + (lane & 15);
  int n = col >> 14, p = col & (HW - 1);
  int khalf = (lane < 16) ? 0 : 16;
  v8f c = {};
  const unsigned short* ap = Wp + (oct * 4) * 512 + lane * 16;
  const unsigned short* bp = AO + col * DIM + khalf;
#pragma unroll
  for (int kt = 0; kt < 4; ++kt){
    v16bf a = *(const v16bf*)(ap + kt * 512);
    v16bf b = *(const v16bf*)(bp + kt * 32);
    c = wmma_bf16(a, b, c);
  }
  int mbase = oct * 16 + ((lane < 16) ? 0 : 8);
#pragma unroll
  for (int r = 0; r < 8; ++r){
    int oc = mbase + r;
    out[(size_t)(n * DIM + oc) * HW + p] = c[r] + bias[oc];
  }
}

// ---------------- host launch ----------------
extern "C" void kernel_launch(void* const* d_in, const int* in_sizes, int n_in,
                              void* d_out, int out_size, void* d_ws, size_t ws_size,
                              hipStream_t stream) {
  const float* x     = (const float*)d_in[0];
  const float* qkv_r = (const float*)d_in[1];
  const float* qkv_i = (const float*)d_in[2];
  const float* qkv_j = (const float*)d_in[3];
  const float* qkv_k = (const float*)d_in[4];
  const float* qkv_b = (const float*)d_in[5];
  const float* dw_r  = (const float*)d_in[6];
  const float* dw_i  = (const float*)d_in[7];
  const float* dw_j  = (const float*)d_in[8];
  const float* dw_k  = (const float*)d_in[9];
  const float* dw_b  = (const float*)d_in[10];
  const float* po_r  = (const float*)d_in[11];
  const float* po_i  = (const float*)d_in[12];
  const float* po_j  = (const float*)d_in[13];
  const float* po_k  = (const float*)d_in[14];
  const float* po_b  = (const float*)d_in[15];
  const float* temp  = (const float*)d_in[16];
  float* out = (float*)d_out;

  unsigned char* w = (unsigned char*)d_ws;
  size_t off = 0;
  auto alloc = [&](size_t bytes) -> void* {
    void* p = w + off;
    off = (off + bytes + 255) & ~(size_t)255;
    return p;
  };
  unsigned short* WQ   = (unsigned short*)alloc(24 * 4 * 512 * 2);        // 96 KB
  unsigned short* WDW  = (unsigned short*)alloc((size_t)9*24*12*512 * 2); // 2.65 MB
  unsigned short* WPO  = (unsigned short*)alloc(8 * 4 * 512 * 2);         // 32 KB
  unsigned short* XBF  = (unsigned short*)alloc((size_t)PTOT * DIM * 2);  // 8.4 MB
  unsigned short* QKV1 = (unsigned short*)alloc((size_t)PTOT * C3  * 2);  // 25.2 MB
  float*          DWO  = (float*)alloc((size_t)PTOT * C3 * 4);            // 50.3 MB
  float*          NRM  = (float*)alloc(512 * 4);
  float*          ATM  = (float*)alloc(8 * 1024 * 4);
  unsigned short* AOUT = (unsigned short*)alloc((size_t)PTOT * DIM * 2);  // 8.4 MB
  (void)ws_size; (void)in_sizes; (void)n_in; (void)out_size;

  // weight packing + activation conversion
  k_pack_1x1<<<192, 256, 0, stream>>>(qkv_r, qkv_i, qkv_j, qkv_k, WQ, 24, 4, 96, 32);
  k_pack_3x3<<<5184, 256, 0, stream>>>(dw_r, dw_i, dw_j, dw_k, WDW);
  k_pack_1x1<<<64, 256, 0, stream>>>(po_r, po_i, po_j, po_k, WPO, 8, 4, 32, 32);
  k_x2bf<<<16384, 256, 0, stream>>>(x, XBF);

  // conv pipeline (WMMA bf16, f32 accumulate)
  k_conv_qkv<<<6144, 256, 0, stream>>>(XBF, WQ, qkv_b, QKV1);
  k_conv_dw<<<6144, 256, 0, stream>>>(QKV1, WDW, dw_b, DWO);

  // channel attention
  k_norms<<<16, 256, 0, stream>>>(DWO, NRM);
  k_attn<<<8, 256, 0, stream>>>(DWO, NRM, temp, ATM);
  k_attnout<<<16384, 256, 0, stream>>>(DWO, ATM, AOUT);

  // output projection
  k_conv_po<<<2048, 256, 0, stream>>>(AOUT, WPO, po_b, out);
}
